// StandardAttention_17884243821432
// MI455X (gfx1250) — matrix operand order repaired and checked by reading, not compiled, not run on hardware
//
#include <hip/hip_runtime.h>

// ---------------------------------------------------------------------------
// MHA forward, B=4 S=2048 D=1024 NH=16 HD=64, gfx1250 (wave32, bf16 WMMA).
//   0) f32 -> bf16 pre-conversion of hidden states + weights (memory bound)
//   1) Q/K/V projections: bf16 WMMA GEMM, 128x128 tile, double-buffered
//      async global->LDS staging (ASYNCcnt ping-pong)
//   2) flash attention, online softmax, bf16 WMMA, no S x S materialization
//   3) output projection -> f32
// ---------------------------------------------------------------------------

typedef __bf16 bf16_t;
typedef __attribute__((ext_vector_type(2)))  __bf16 v2bf;
typedef __attribute__((ext_vector_type(8)))  __bf16 v8bf;
typedef __attribute__((ext_vector_type(16))) __bf16 v16bf;
typedef __attribute__((ext_vector_type(8)))  float  v8f;

#define BATCH 4
#define SEQ   2048
#define BDIM  1024
#define NHEAD 16
#define HDIM  64

static __device__ __forceinline__ v16bf cat8(v8bf lo, v8bf hi) {
  return __builtin_shufflevector(lo, hi, 0,1,2,3,4,5,6,7,8,9,10,11,12,13,14,15);
}

// A-fragment 16x32 bf16 (ISA 7.12.2): lanes 0-15 rows 0-15 hold K 0..7 / 16..23,
// lanes 16-31 same rows hold K 8..15 / 24..31.  p -> [row][k], row stride `stride`.
static __device__ __forceinline__ v16bf load_a_frag(const bf16_t* p, int stride, int lane) {
  const int r = lane & 15, h = lane >> 4;
  const bf16_t* row = p + r * stride + 8 * h;
  v8bf lo = *(const v8bf*)(row);
  v8bf hi = *(const v8bf*)(row + 16);
  return cat8(lo, hi);
}

// B-fragment 32x16 bf16: lanes 0-15 col N=lane K 0..7 / 16..23, lanes 16-31 K 8..15 / 24..31.
// p -> [n][k] (K-contiguous) storage with n stride `stride`.
static __device__ __forceinline__ v16bf load_b_frag(const bf16_t* p, int stride, int lane) {
  const int n = lane & 15, h = lane >> 4;
  const bf16_t* col = p + n * stride + 8 * h;
  v8bf lo = *(const v8bf*)(col);
  v8bf hi = *(const v8bf*)(col + 16);
  return cat8(lo, hi);
}

static __device__ __forceinline__ v8f wmma_bf16(v16bf a, v16bf b, v8f c) {
  return __builtin_amdgcn_wmma_f32_16x16x32_bf16(false, a, false, b, (short)0, c,
                                                 false, false);
}

// Async global -> LDS 16B copy (CDNA5, ASYNCcnt-tracked, bypasses VGPRs).
// VDST VGPR carries the wave-relative LDS byte offset = low 32 bits of the
// flat shared-memory pointer (ISA 10.2 aperture mapping).
static __device__ __forceinline__ void async_copy_b128(void* lds_dst, const void* g_src) {
  unsigned lds_off = (unsigned)(size_t)lds_dst;
  asm volatile("global_load_async_to_lds_b128 %0, %1, off"
               :: "v"(lds_off), "v"(g_src) : "memory");
}
static __device__ __forceinline__ void wait_async0() {
  asm volatile("s_wait_asynccnt 0x0" ::: "memory");
}

// ---------------------------------------------------------------------------
// f32 -> bf16 bulk convert (n multiple of 2048)
// ---------------------------------------------------------------------------
__global__ __launch_bounds__(256)
void cvt_f32_to_bf16(const float* __restrict__ in, bf16_t* __restrict__ out, int n) {
  const int i = (blockIdx.x * 256 + threadIdx.x) * 8;
  if (i + 8 <= n) {
    float4 a = *(const float4*)(in + i);
    float4 b = *(const float4*)(in + i + 4);
    v8bf o;
    o[0] = (bf16_t)a.x; o[1] = (bf16_t)a.y; o[2] = (bf16_t)a.z; o[3] = (bf16_t)a.w;
    o[4] = (bf16_t)b.x; o[5] = (bf16_t)b.y; o[6] = (bf16_t)b.z; o[7] = (bf16_t)b.w;
    *(v8bf*)(out + i) = o;
  }
}

// ---------------------------------------------------------------------------
// O[M,N] = X[M,K](bf16) @ W[N,K](bf16)^T + bias[N](f32)
// 256 threads = 8 waves, tile 128(M) x 128(N), BK = 32, double-buffered async
// LDS staging.  Wave (wm,wn) = (w>>1, w&1) computes 32(M) x 64(N).
// ---------------------------------------------------------------------------
template <bool OUT_BF16>
__global__ __launch_bounds__(256)
void gemm_bf16(const bf16_t* __restrict__ X, const bf16_t* __restrict__ W,
               const float* __restrict__ bias, void* __restrict__ Ov,
               int M, int N, int K) {
  __shared__ bf16_t lA[2][128][40];   // [m][k] ping-pong, stride 40 keeps 16B align
  __shared__ bf16_t lB[2][128][40];   // [n][k] == direct copy of W rows

  const int tid  = threadIdx.x;
  const int lane = tid & 31;
  const int w    = tid >> 5;
  const int wm   = w >> 1;          // 0..3 : 32-row quadrant
  const int wn   = w & 1;           // 0..1 : 64-col half
  const int m0   = blockIdx.x * 128;
  const int n0   = blockIdx.y * 128;

  v8f acc[2][4] = {};

  const int srow = tid >> 1;          // 0..127
  const int sseg = (tid & 1) * 16;    // 0 or 16 : 16 elems (2 x b128) per array

  const bf16_t* asrc = X + (size_t)(m0 + srow) * K + sseg;
  const bf16_t* bsrc = W + (size_t)(n0 + srow) * K + sseg;

  // prologue: stage tile k0=0 into buffer 0
  async_copy_b128(&lA[0][srow][sseg],     asrc);
  async_copy_b128(&lA[0][srow][sseg + 8], asrc + 8);
  async_copy_b128(&lB[0][srow][sseg],     bsrc);
  async_copy_b128(&lB[0][srow][sseg + 8], bsrc + 8);

  int cur = 0;
  for (int k0 = 0; k0 < K; k0 += 32) {
    wait_async0();       // drain tile issued one full compute phase ago
    __syncthreads();     // all waves done staging cur / done reading cur^1

    const int nxt = cur ^ 1;
    if (k0 + 32 < K) {   // issue next tile; overlaps the WMMAs below
      const bf16_t* an = asrc + k0 + 32;
      const bf16_t* bn = bsrc + k0 + 32;
      async_copy_b128(&lA[nxt][srow][sseg],     an);
      async_copy_b128(&lA[nxt][srow][sseg + 8], an + 8);
      async_copy_b128(&lB[nxt][srow][sseg],     bn);
      async_copy_b128(&lB[nxt][srow][sseg + 8], bn + 8);
    }

    v16bf a0 = load_a_frag(&lA[cur][wm * 32][0],      40, lane);
    v16bf a1 = load_a_frag(&lA[cur][wm * 32 + 16][0], 40, lane);
    #pragma unroll
    for (int t = 0; t < 4; ++t) {
      v16bf b = load_b_frag(&lB[cur][wn * 64 + t * 16][0], 40, lane);
      acc[0][t] = wmma_bf16(a0, b, acc[0][t]);
      acc[1][t] = wmma_bf16(a1, b, acc[1][t]);
    }
    cur = nxt;
  }

  const int r = lane & 15, h = lane >> 4;
  #pragma unroll
  for (int sub = 0; sub < 2; ++sub) {
    #pragma unroll
    for (int t = 0; t < 4; ++t) {
      const int n  = n0 + wn * 64 + t * 16 + r;
      const float bv = bias[n];
      #pragma unroll
      for (int j = 0; j < 8; ++j) {
        const int mrow = m0 + wm * 32 + sub * 16 + j + 8 * h;
        const float val = acc[sub][t][j] + bv;
        if (OUT_BF16) ((bf16_t*)Ov)[(size_t)mrow * N + n] = (bf16_t)val;
        else          ((float*)Ov)[(size_t)mrow * N + n]  = val;
      }
    }
  }
}

// ---------------------------------------------------------------------------
// Flash attention: grid = (S/64, B*NH), 128 threads (4 waves), each wave owns
// 16 query rows, KV in 64-wide steps with online softmax.  bf16 [B*S, D].
// ---------------------------------------------------------------------------
__global__ __launch_bounds__(128)
void flash_attn(const bf16_t* __restrict__ Q, const bf16_t* __restrict__ Kd,
                const bf16_t* __restrict__ Vd, const float* __restrict__ mask,
                bf16_t* __restrict__ ctx) {
  __shared__ bf16_t lK[64][72];        // [kv][hd] -> B for Q.K^T
  __shared__ bf16_t lV[64][72];        // [hd][kv] -> B for P.V (transposed)
  __shared__ bf16_t lP[4][16][72];     // per-wave P tile [row][kv]

  const int tid  = threadIdx.x;
  const int lane = tid & 31;
  const int w    = tid >> 5;
  const int b    = blockIdx.y / NHEAD;
  const int hID  = blockIdx.y % NHEAD;
  const int q0   = blockIdx.x * 64;

  const size_t rowQ = (size_t)b * SEQ + q0 + w * 16;
  const bf16_t* qp  = Q + rowQ * BDIM + hID * HDIM;
  const v16bf qa0 = load_a_frag(qp,      BDIM, lane);   // hd 0..31
  const v16bf qa1 = load_a_frag(qp + 32, BDIM, lane);   // hd 32..63

  v8f oacc[4] = {};
  float m[8], l[8];
  #pragma unroll
  for (int j = 0; j < 8; ++j) { m[j] = -3.0e38f; l[j] = 0.0f; }

  // K staging: 2 threads per kv row, 32 bf16 each (4 async b128)
  const int krow = tid >> 1, kseg = (tid & 1) * 32;
  // V staging: thread handles 2 kv rows x 16 hd, packs v2bf column stores
  const int vp = tid >> 2, vseg = (tid & 3) * 16;

  for (int kv0 = 0; kv0 < SEQ; kv0 += 64) {
    const size_t kvBase = (size_t)b * SEQ + kv0;
    const bf16_t* ksrc = Kd + (kvBase + krow) * BDIM + hID * HDIM + kseg;
    #pragma unroll
    for (int i = 0; i < 4; ++i)
      async_copy_b128(&lK[krow][kseg + i * 8], ksrc + i * 8);

    // V transpose via VGPRs (overlaps the async K copies)
    const bf16_t* vsrc = Vd + (kvBase + 2 * vp) * BDIM + hID * HDIM + vseg;
    {
      v8bf r0a = *(const v8bf*)(vsrc);
      v8bf r0b = *(const v8bf*)(vsrc + 8);
      v8bf r1a = *(const v8bf*)(vsrc + BDIM);
      v8bf r1b = *(const v8bf*)(vsrc + BDIM + 8);
      #pragma unroll
      for (int e = 0; e < 8; ++e) {
        v2bf t0; t0[0] = r0a[e]; t0[1] = r1a[e];
        *(v2bf*)&lV[vseg + e][2 * vp] = t0;
        v2bf t1; t1[0] = r0b[e]; t1[1] = r1b[e];
        *(v2bf*)&lV[vseg + 8 + e][2 * vp] = t1;
      }
    }

    // prefetch next KV tile while this one is consumed
    if (kv0 + 64 < SEQ) {
      __builtin_prefetch(Kd + (kvBase + 64 + krow) * BDIM + hID * HDIM + kseg, 0, 1);
      __builtin_prefetch(Vd + (kvBase + 64 + 2 * vp) * BDIM + hID * HDIM + vseg, 0, 1);
    }

    wait_async0();
    __syncthreads();

    // --- scores: S = (Q.K^T)/8 + mask ---
    v8f st[4];
    #pragma unroll
    for (int t = 0; t < 4; ++t) {
      v8f z = {};
      v16bf b0 = load_b_frag(&lK[t * 16][0],  72, lane);
      z = wmma_bf16(qa0, b0, z);
      v16bf b1 = load_b_frag(&lK[t * 16][32], 72, lane);
      z = wmma_bf16(qa1, b1, z);
      const float mt =
          (1.0f - mask[(size_t)b * SEQ + kv0 + t * 16 + (lane & 15)]) * -10000.0f;
      #pragma unroll
      for (int j = 0; j < 8; ++j) z[j] = z[j] * 0.125f + mt;
      st[t] = z;
    }

    // --- online softmax: row (j, lane>>4) lives in one 16-lane half ---
    #pragma unroll
    for (int j = 0; j < 8; ++j) {
      float rm = fmaxf(fmaxf(st[0][j], st[1][j]), fmaxf(st[2][j], st[3][j]));
      rm = fmaxf(rm, __shfl_xor(rm, 1, 32));
      rm = fmaxf(rm, __shfl_xor(rm, 2, 32));
      rm = fmaxf(rm, __shfl_xor(rm, 4, 32));
      rm = fmaxf(rm, __shfl_xor(rm, 8, 32));
      const float mn    = fmaxf(m[j], rm);
      const float alpha = __expf(m[j] - mn);
      m[j] = mn;
      float rs = 0.0f;
      #pragma unroll
      for (int t = 0; t < 4; ++t) {
        const float p = __expf(st[t][j] - mn);
        st[t][j] = p;
        rs += p;
      }
      rs += __shfl_xor(rs, 1, 32);
      rs += __shfl_xor(rs, 2, 32);
      rs += __shfl_xor(rs, 4, 32);
      rs += __shfl_xor(rs, 8, 32);
      l[j] = l[j] * alpha + rs;
      #pragma unroll
      for (int t = 0; t < 4; ++t) oacc[t][j] *= alpha;
    }

    // --- P (C layout) -> LDS -> A fragments (same-wave DS ops are in-order) ---
    {
      const int r = lane & 15, h = lane >> 4;
      #pragma unroll
      for (int t = 0; t < 4; ++t)
        #pragma unroll
        for (int j = 0; j < 8; ++j)
          lP[w][j + 8 * h][t * 16 + r] = (bf16_t)st[t][j];
    }
    __builtin_amdgcn_wave_barrier();

    const v16bf pa0 = load_a_frag(&lP[w][0][0],  72, lane);  // kv 0..31
    const v16bf pa1 = load_a_frag(&lP[w][0][32], 72, lane);  // kv 32..63

    // --- O += P.V ---
    #pragma unroll
    for (int t = 0; t < 4; ++t) {
      v16bf vb0 = load_b_frag(&lV[t * 16][0],  72, lane);
      oacc[t] = wmma_bf16(pa0, vb0, oacc[t]);
      v16bf vb1 = load_b_frag(&lV[t * 16][32], 72, lane);
      oacc[t] = wmma_bf16(pa1, vb1, oacc[t]);
    }
    __syncthreads();
  }

  const int r = lane & 15, h = lane >> 4;
  #pragma unroll
  for (int j = 0; j < 8; ++j) {
    const float inv = 1.0f / l[j];
    const size_t mrow = rowQ + j + 8 * h;
    #pragma unroll
    for (int t = 0; t < 4; ++t)
      ctx[mrow * BDIM + hID * HDIM + t * 16 + r] = (bf16_t)(oacc[t][j] * inv);
  }
}

// ---------------------------------------------------------------------------
extern "C" void kernel_launch(void* const* d_in, const int* in_sizes, int n_in,
                              void* d_out, int out_size, void* d_ws, size_t ws_size,
                              hipStream_t stream) {
  (void)in_sizes; (void)n_in; (void)out_size; (void)ws_size;
  const float* hs   = (const float*)d_in[0];
  const float* mask = (const float*)d_in[1];
  const float* Wq   = (const float*)d_in[2];
  const float* bq   = (const float*)d_in[3];
  const float* Wk   = (const float*)d_in[4];
  const float* bk   = (const float*)d_in[5];
  const float* Wv   = (const float*)d_in[6];
  const float* bv   = (const float*)d_in[7];
  const float* Wo   = (const float*)d_in[8];
  const float* bo   = (const float*)d_in[9];

  const int M  = BATCH * SEQ;            // 8192
  const size_t MD = (size_t)M * BDIM;    // 8M elements
  const size_t DD = (size_t)BDIM * BDIM; // 1M elements

  bf16_t* Qw  = (bf16_t*)d_ws;
  bf16_t* Kw  = Qw  + MD;
  bf16_t* Vw  = Kw  + MD;
  bf16_t* Cw  = Vw  + MD;
  bf16_t* hsb = Cw  + MD;
  bf16_t* Wqb = hsb + MD;
  bf16_t* Wkb = Wqb + DD;
  bf16_t* Wvb = Wkb + DD;
  bf16_t* Wob = Wvb + DD;                // total 88 MB of workspace

  // 0) bulk f32 -> bf16 conversion (memory bound, ~3us at 23.3 TB/s)
  cvt_f32_to_bf16<<<(int)(MD / 2048), 256, 0, stream>>>(hs, hsb, (int)MD);
  cvt_f32_to_bf16<<<(int)(DD / 2048), 256, 0, stream>>>(Wq, Wqb, (int)DD);
  cvt_f32_to_bf16<<<(int)(DD / 2048), 256, 0, stream>>>(Wk, Wkb, (int)DD);
  cvt_f32_to_bf16<<<(int)(DD / 2048), 256, 0, stream>>>(Wv, Wvb, (int)DD);
  cvt_f32_to_bf16<<<(int)(DD / 2048), 256, 0, stream>>>(Wo, Wob, (int)DD);

  // 1) projections
  dim3 gg(M / 128, BDIM / 128);
  gemm_bf16<true><<<gg, 256, 0, stream>>>(hsb, Wqb, bq, Qw, M, BDIM, BDIM);
  gemm_bf16<true><<<gg, 256, 0, stream>>>(hsb, Wkb, bk, Kw, M, BDIM, BDIM);
  gemm_bf16<true><<<gg, 256, 0, stream>>>(hsb, Wvb, bv, Vw, M, BDIM, BDIM);

  // 2) attention
  flash_attn<<<dim3(SEQ / 64, BATCH * NHEAD), 128, 0, stream>>>(Qw, Kw, Vw, mask, Cw);

  // 3) output projection -> f32
  gemm_bf16<false><<<gg, 256, 0, stream>>>(Cw, Wob, bo, (float*)d_out, M, BDIM, BDIM);
}
